// MultiTimeAttention_31473520345656
// MI455X (gfx1250) — matrix operand order repaired and checked by reading, not compiled, not run on hardware
//
#include <hip/hip_runtime.h>
#include <hip/hip_bf16.h>

// ---------------------------------------------------------------------------
// MultiTimeAttention for MI455X (gfx1250, wave32, WMMA bf16 16x16x32,
// async global->LDS staging, fragment-friendly operand layouts)
// B=8, LQ=128, LK=256, E=128, H=8, dk=16, D=64, NHID=128
// ---------------------------------------------------------------------------

typedef __bf16 bf16_t;
typedef __attribute__((ext_vector_type(16))) __bf16 v16bf;
typedef __attribute__((ext_vector_type(8)))  __bf16 v8bf;
typedef __attribute__((ext_vector_type(8)))  float  v8f;
typedef int v4i __attribute__((vector_size(16)));   // matches builtin pointee

#define WMMA_BF16(a, b, c) \
  __builtin_amdgcn_wmma_f32_16x16x32_bf16(false, (a), false, (b), (short)0, (c), false, false)

// --- async global->LDS copy (CDNA5), with portable fallback -----------------
#if defined(__has_builtin)
#if __has_builtin(__builtin_amdgcn_global_load_async_to_lds_b128)
#define ASYNC_LDS 1
#endif
#endif
#ifndef ASYNC_LDS
#define ASYNC_LDS 0
#endif

__device__ __forceinline__ void cp16_g2l(bf16_t* lds, const bf16_t* g) {
#if ASYNC_LDS
  __builtin_amdgcn_global_load_async_to_lds_b128(
      (__attribute__((address_space(1))) v4i*)(unsigned long long)g,
      (__attribute__((address_space(3))) v4i*)(unsigned int)(unsigned long long)lds,
      0, 0);
#else
  *(v8bf*)lds = *(const v8bf*)g;
#endif
}

__device__ __forceinline__ void cp_commit_wait() {
#if ASYNC_LDS
#if __has_builtin(__builtin_amdgcn_s_wait_asynccnt)
  __builtin_amdgcn_s_wait_asynccnt(0);
#else
  asm volatile("s_wait_asynccnt 0x0" ::: "memory");
#endif
#endif
}

// --- fragment gathers from LDS (contiguous -> ds_load_b128 pairs) -----------
// A tile in LDS: [16 m][32 k] row-major.
// lane<16: M=lane, K {0..7}u{16..23}; lane>=16: M=lane-16, K {8..15}u{24..31}
__device__ __forceinline__ v16bf ld_frag_a(const bf16_t* s) {
  const int lane = threadIdx.x & 31;
  const bf16_t* p = s + (lane & 15) * 32 + ((lane & 16) ? 8 : 0);
  v8bf lo = *(const v8bf*)p;          // K kb..kb+7
  v8bf hi = *(const v8bf*)(p + 16);   // K kb+16..kb+23
  return __builtin_shufflevector(lo, hi, 0,1,2,3,4,5,6,7,8,9,10,11,12,13,14,15);
}
// B tile in LDS: transposed [16 n][32 k] row-major.
// lane<16: N=lane, K {0..7}u{16..23}; lane>=16: N=lane-16, K {8..15}u{24..31}
__device__ __forceinline__ v16bf ld_frag_bt(const bf16_t* s) {
  const int lane = threadIdx.x & 31;
  const bf16_t* p = s + (lane & 15) * 32 + ((lane & 16) ? 8 : 0);
  v8bf lo = *(const v8bf*)p;
  v8bf hi = *(const v8bf*)(p + 16);
  return __builtin_shufflevector(lo, hi, 0,1,2,3,4,5,6,7,8,9,10,11,12,13,14,15);
}
// C/D (16x16 f32): element j of lane L -> M = j + 8*(L>=16), N = L&15

// ---------------------------------------------------------------------------
// Prep kernels: lay operands out as bf16 (A: [M][K]; B: transposed [N][K])
// ---------------------------------------------------------------------------
__global__ void convert_bf16_kernel(bf16_t* __restrict__ dst,
                                    const float* __restrict__ src, int n) {
  for (int i = blockIdx.x * blockDim.x + threadIdx.x; i < n;
       i += gridDim.x * blockDim.x)
    dst[i] = (bf16_t)src[i];
}

__global__ void transpose_bf16_kernel(bf16_t* __restrict__ dst,
                                      const float* __restrict__ src, int R, int C) {
  int i = blockIdx.x * blockDim.x + threadIdx.x;
  if (i < R * C) {
    int r = i / C, c = i % C;
    dst[c * R + r] = (bf16_t)src[i];
  }
}

// MVT[b][d][k] = mask ? value : 0 ; MFT[b][d][k] = mask ? 1 : 0  (d-major)
__global__ void maskv_kernel(bf16_t* __restrict__ MVT, bf16_t* __restrict__ MFT,
                             const float* __restrict__ V, const int* __restrict__ Mk) {
  int i = blockIdx.x * blockDim.x + threadIdx.x;  // over [8][256][64]
  if (i < 8 * 256 * 64) {
    int b = i >> 14, k = (i >> 6) & 255, d = i & 63;
    int m = Mk[i];
    int o = (b * 64 + d) * 256 + k;
    MVT[o] = (bf16_t)(m ? V[i] : 0.f);
    MFT[o] = (bf16_t)(m ? 1.f : 0.f);
  }
}

// ---------------------------------------------------------------------------
// Generic GEMM: C[M,N] = A[M,K] @ Bt[N,K]^T + bias  (bf16 in, f32 acc)
// 256 threads = 8 waves; 64x64 block tile; each wave two 16x16 tiles.
// Requires M%64==0, N%64==0, K%32==0.
// ---------------------------------------------------------------------------
template <bool OUT_F32>
__global__ __launch_bounds__(256) void gemm_bf16_kernel(
    const bf16_t* __restrict__ A, const bf16_t* __restrict__ Bt,
    const float* __restrict__ bias, void* __restrict__ Cout,
    int M, int N, int K) {
  __shared__ bf16_t As[64 * 32];
  __shared__ bf16_t Bs[64 * 32];
  const int tid = threadIdx.x;
  const int wave = tid >> 5;
  const int bm = blockIdx.y * 64, bn = blockIdx.x * 64;
  const int tm = wave >> 1, tn0 = (wave & 1) * 2;
  const int r = tid >> 2, c8 = (tid & 3) * 8;  // 16B chunk per thread

  v8f acc[2] = {};
  for (int k0 = 0; k0 < K; k0 += 32) {
    __syncthreads();
    cp16_g2l(As + r * 32 + c8, A + (bm + r) * K + k0 + c8);
    cp16_g2l(Bs + r * 32 + c8, Bt + (bn + r) * K + k0 + c8);
    cp_commit_wait();
    __syncthreads();
    v16bf a = ld_frag_a(As + tm * 16 * 32);
#pragma unroll
    for (int t = 0; t < 2; ++t) {
      v16bf b = ld_frag_bt(Bs + (tn0 + t) * 16 * 32);
      acc[t] = WMMA_BF16(a, b, acc[t]);
    }
  }
  const int lane = tid & 31;
  const int n = lane & 15;
  const int hi8 = (lane & 16) ? 8 : 0;
#pragma unroll
  for (int t = 0; t < 2; ++t) {
    const int col = bn + (tn0 + t) * 16 + n;
    const float bv = bias[col];
#pragma unroll
    for (int j = 0; j < 8; ++j) {
      const int row = bm + tm * 16 + j + hi8;
      const float v = acc[t][j] + bv;
      if (OUT_F32) ((float*)Cout)[row * N + col] = v;
      else         ((bf16_t*)Cout)[row * N + col] = (bf16_t)v;
    }
  }
}

// ---------------------------------------------------------------------------
// Scores + exp: block = (b, h, 16-row q tile).
// s = Q_h[16,16] @ K_h^T[16,256] * 0.25 (dk padded 16->32 with real zeros),
// e = exp(s - rowmax) -> EB[b,h,128,256] bf16.
// ---------------------------------------------------------------------------
__global__ __launch_bounds__(256) void scores_exp_kernel(
    const bf16_t* __restrict__ Qp, const bf16_t* __restrict__ Kp,
    bf16_t* __restrict__ EB) {
  __shared__ bf16_t Qs[16 * 32];    // [m][k], k 16..31 zero
  __shared__ bf16_t Ks[256 * 32];   // transposed [key][k], k 16..31 zero
  __shared__ float  Ss[16 * 256];
  __shared__ float  partial[16 * 16];
  __shared__ float  rmax[16];
  const int tid = threadIdx.x;
  const int qt = blockIdx.x & 7;
  const int h  = (blockIdx.x >> 3) & 7;
  const int b  = blockIdx.x >> 6;

  // zero the dk padding (cols 16..31)
  v8bf z = {};
  if (tid < 32) {
    int rr = tid >> 1, cc = 16 + (tid & 1) * 8;
    *(v8bf*)(Qs + rr * 32 + cc) = z;
  }
#pragma unroll
  for (int i = 0; i < 2; ++i) {
    int ch = tid + i * 256;
    int key = ch >> 1, cc = 16 + (ch & 1) * 8;
    *(v8bf*)(Ks + key * 32 + cc) = z;
  }
  // async-stage the real data (cols 0..15)
  if (tid < 32) {
    int rr = tid >> 1, cc = (tid & 1) * 8;
    cp16_g2l(Qs + rr * 32 + cc,
             Qp + (b * 128 + qt * 16 + rr) * 128 + h * 16 + cc);
  }
#pragma unroll
  for (int i = 0; i < 2; ++i) {
    int ch = tid + i * 256;
    int key = ch >> 1, cc = (ch & 1) * 8;
    cp16_g2l(Ks + key * 32 + cc,
             Kp + (b * 256 + key) * 128 + h * 16 + cc);
  }
  cp_commit_wait();
  __syncthreads();

  const int wave = tid >> 5;
  const int lane = tid & 31;
  const int n = lane & 15;
  const int hi8 = (lane & 16) ? 8 : 0;
  v16bf a = ld_frag_a(Qs);
#pragma unroll
  for (int t = 0; t < 2; ++t) {          // each wave: 2 key tiles of 16
    const int kt = wave * 2 + t;
    v16bf bk = ld_frag_bt(Ks + kt * 16 * 32);
    v8f acc = {};
    acc = WMMA_BF16(a, bk, acc);
#pragma unroll
    for (int j = 0; j < 8; ++j)
      Ss[(j + hi8) * 256 + kt * 16 + n] = acc[j] * 0.25f;  // 1/sqrt(dk)
  }
  __syncthreads();

  {  // row max over 256 keys
    const int rr = tid >> 4, t = tid & 15;
    float pm = -3.0e38f;
    for (int c = t; c < 256; c += 16) pm = fmaxf(pm, Ss[rr * 256 + c]);
    partial[rr * 16 + t] = pm;
  }
  __syncthreads();
  if (tid < 16) {
    float m = -3.0e38f;
#pragma unroll
    for (int t = 0; t < 16; ++t) m = fmaxf(m, partial[tid * 16 + t]);
    rmax[tid] = m;
  }
  __syncthreads();

  bf16_t* Eout = EB + ((b * 8 + h) * 128 + qt * 16) * 256;
#pragma unroll
  for (int i = 0; i < 16; ++i) {
    int lin = tid + i * 256;
    int rr = lin >> 8, cc = lin & 255;
    Eout[rr * 256 + cc] = (bf16_t)expf(Ss[rr * 256 + cc] - rmax[rr]);
  }
}

// ---------------------------------------------------------------------------
// Attention num/den GEMMs + merge: block = (b,h).
// num = E[128,256] @ (m*V)[256,64]; den = E @ m; out = num/den -> XmB bf16.
// ---------------------------------------------------------------------------
__global__ __launch_bounds__(256) void attn_kernel(
    const bf16_t* __restrict__ EB, const bf16_t* __restrict__ MVT,
    const bf16_t* __restrict__ MFT, bf16_t* __restrict__ XmB) {
  __shared__ bf16_t Es[128 * 32];   // [q][k]
  __shared__ bf16_t MVs[64 * 32];   // transposed [d][k]
  __shared__ bf16_t MFs[64 * 32];   // transposed [d][k]
  const int tid = threadIdx.x;
  const int h = blockIdx.x & 7;
  const int b = blockIdx.x >> 3;
  const bf16_t* E  = EB  + (b * 8 + h) * 128 * 256;
  const bf16_t* MV = MVT + b * 64 * 256;
  const bf16_t* MF = MFT + b * 64 * 256;
  const int wave = tid >> 5;                  // M tile: 8 x 16 rows = 128
  const int r = tid >> 2, c8 = (tid & 3) * 8; // 16B chunk for 64-row tiles

  v8f accn[4] = {};
  v8f accd[4] = {};
  for (int k0 = 0; k0 < 256; k0 += 32) {
    __syncthreads();
#pragma unroll
    for (int i = 0; i < 2; ++i) {  // Es: 128x32 -> 512 chunks
      int ch = tid + i * 256;
      int rr = ch >> 2, cc = (ch & 3) * 8;
      cp16_g2l(Es + rr * 32 + cc, E + rr * 256 + k0 + cc);
    }
    cp16_g2l(MVs + r * 32 + c8, MV + r * 256 + k0 + c8);
    cp16_g2l(MFs + r * 32 + c8, MF + r * 256 + k0 + c8);
    cp_commit_wait();
    __syncthreads();
    v16bf a = ld_frag_a(Es + wave * 16 * 32);
#pragma unroll
    for (int tn = 0; tn < 4; ++tn) {
      v16bf bv = ld_frag_bt(MVs + tn * 16 * 32);
      accn[tn] = WMMA_BF16(a, bv, accn[tn]);
      v16bf bf = ld_frag_bt(MFs + tn * 16 * 32);
      accd[tn] = WMMA_BF16(a, bf, accd[tn]);
    }
  }
  const int lane = tid & 31;
  const int n = lane & 15;
  const int hi8 = (lane & 16) ? 8 : 0;
#pragma unroll
  for (int tn = 0; tn < 4; ++tn) {
#pragma unroll
    for (int j = 0; j < 8; ++j) {
      const int row = wave * 16 + j + hi8;
      const float num = accn[tn][j];
      const float den = accd[tn][j];
      XmB[(b * 128 + row) * 512 + h * 64 + tn * 16 + n] =
          (bf16_t)(num / fmaxf(den, 1e-30f));
    }
  }
}

// ---------------------------------------------------------------------------
extern "C" void kernel_launch(void* const* d_in, const int* in_sizes, int n_in,
                              void* d_out, int out_size, void* d_ws, size_t ws_size,
                              hipStream_t stream) {
  (void)in_sizes; (void)n_in; (void)out_size; (void)ws_size;
  const float* query = (const float*)d_in[0];  // [8,128,128]
  const float* key_  = (const float*)d_in[1];  // [8,256,128]
  const float* value = (const float*)d_in[2];  // [8,256,64]
  const int*   mask  = (const int*)d_in[3];    // [8,256,64]
  const float* Wq = (const float*)d_in[4];     // [128,128]
  const float* bq = (const float*)d_in[5];     // [128]
  const float* Wk = (const float*)d_in[6];     // [128,128]
  const float* bk = (const float*)d_in[7];     // [128]
  const float* Wo = (const float*)d_in[8];     // [512,128]
  const float* bo = (const float*)d_in[9];     // [128]
  float* out = (float*)d_out;                  // [8,128,128] f32

  // bf16 workspace carve-out (~7.5 MB, lives in L2)
  bf16_t* QA  = (bf16_t*)d_ws;          // [1024][128]  A of Qproj
  bf16_t* KA  = QA  + 131072;           // [2048][128]  A of Kproj
  bf16_t* WqT = KA  + 262144;           // [128][128]   Bt of Qproj
  bf16_t* WkT = WqT + 16384;            // [128][128]   Bt of Kproj
  bf16_t* WoT = WkT + 16384;            // [128][512]   Bt of out GEMM
  bf16_t* MVT = WoT + 65536;            // [8][64][256] Bt of attn (m*V)
  bf16_t* MFT = MVT + 131072;           // [8][64][256] Bt of attn (m)
  bf16_t* QpB = MFT + 131072;           // [1024][128]  q projected
  bf16_t* KpB = QpB + 131072;           // [2048][128]  k projected
  bf16_t* EB  = KpB + 262144;           // [64][128][256] exp(scores)
  bf16_t* XmB = EB  + 2097152;          // [1024][512]  merged heads

  const dim3 blk(256);
  convert_bf16_kernel<<<256, blk, 0, stream>>>(QA, query, 131072);
  convert_bf16_kernel<<<256, blk, 0, stream>>>(KA, key_, 262144);
  transpose_bf16_kernel<<<64, blk, 0, stream>>>(WqT, Wq, 128, 128);
  transpose_bf16_kernel<<<64, blk, 0, stream>>>(WkT, Wk, 128, 128);
  transpose_bf16_kernel<<<256, blk, 0, stream>>>(WoT, Wo, 512, 128);
  maskv_kernel<<<512, blk, 0, stream>>>(MVT, MFT, value, mask);

  gemm_bf16_kernel<false><<<dim3(2, 16), blk, 0, stream>>>(QA, WqT, bq, QpB, 1024, 128, 128);
  gemm_bf16_kernel<false><<<dim3(2, 32), blk, 0, stream>>>(KA, WkT, bk, KpB, 2048, 128, 128);
  scores_exp_kernel<<<dim3(512), blk, 0, stream>>>(QpB, KpB, EB);
  attn_kernel<<<dim3(64), blk, 0, stream>>>(EB, MVT, MFT, XmB);
  gemm_bf16_kernel<true><<<dim3(2, 16), blk, 0, stream>>>(XmB, WoT, bo, out, 1024, 128, 512);
}
